// THAN_one_layer_25958782337225
// MI455X (gfx1250) — matrix operand order repaired and checked by reading, not compiled, not run on hardware
//
#include <hip/hip_runtime.h>
#include <hip/hip_bf16.h>

typedef __attribute__((ext_vector_type(16))) __bf16 v16bf;
typedef __attribute__((ext_vector_type(8)))  float  v8f;

#define Bq   8
#define Nn   2048
#define Dd   64
#define TM   16     // rows per wave tile
#define BJ   32     // j-block (WMMA K)

// float -> bf16 bits, round-to-nearest-even
__device__ __forceinline__ unsigned short f2bf(float f) {
    unsigned u = __float_as_uint(f);
    u = u + 0x7FFFu + ((u >> 16) & 1u);
    return (unsigned short)(u >> 16);
}

union V16 {
    v16bf v;
    uint4 q[2];
};

// ---------------------------------------------------------------------------
// Kernel 1: hw = relu(h @ W) stored transposed [b][d][n] as bf16;
//           t1 = hw @ a1, t2 = hw @ a2 (per-row scalars).
// One wave (32 lanes) per node row; lane owns output cols (lane, lane+32).
// ---------------------------------------------------------------------------
__global__ void gat_prep_kernel(const float* __restrict__ h,
                                const float* __restrict__ W,
                                const float* __restrict__ a1,
                                const float* __restrict__ a2,
                                unsigned short* __restrict__ hwT,
                                float* __restrict__ t1,
                                float* __restrict__ t2) {
    const int lane = threadIdx.x & 31;
    const int w    = threadIdx.x >> 5;
    const int row  = blockIdx.x * 8 + w;          // 0 .. B*N-1
    const int b    = row >> 11;
    const int nidx = row & (Nn - 1);

    const float* __restrict__ hrow = h + (size_t)row * Dd;
    float acc0 = 0.f, acc1 = 0.f;
#pragma unroll 8
    for (int k = 0; k < Dd; ++k) {
        float hk = hrow[k];                        // wave-uniform load
        acc0 = fmaf(hk, W[k * Dd + lane],      acc0);
        acc1 = fmaf(hk, W[k * Dd + lane + 32], acc1);
    }
    acc0 = fmaxf(acc0, 0.f);
    acc1 = fmaxf(acc1, 0.f);

    hwT[((size_t)b * Dd + lane)      * Nn + nidx] = f2bf(acc0);
    hwT[((size_t)b * Dd + lane + 32) * Nn + nidx] = f2bf(acc1);

    float p1 = acc0 * a1[lane] + acc1 * a1[lane + 32];
    float p2 = acc0 * a2[lane] + acc1 * a2[lane + 32];
#pragma unroll
    for (int off = 16; off >= 1; off >>= 1) {
        p1 += __shfl_xor(p1, off, 32);
        p2 += __shfl_xor(p2, off, 32);
    }
    if (lane == 0) {
        t1[row] = p1;
        t2[row] = p2;
    }
}

// ---------------------------------------------------------------------------
// Kernel 2: fused masked-softmax(e) @ hw with bf16 WMMA, + h_root select.
// One wave per (b, 16-row tile); 8 waves / 256-thread block; 128 blocks.
// Two passes over the tile's adj slice (128 KB -> second pass hits L2).
// ---------------------------------------------------------------------------
__global__ void gat_attn_kernel(const float* __restrict__ h0,
                                const int*   __restrict__ adj,
                                const float* __restrict__ h_root,
                                const unsigned short* __restrict__ hwT, // [b][d][n] bf16
                                const float* __restrict__ t1,
                                const float* __restrict__ t2,
                                float* __restrict__ out) {
    // per-wave 16x32 bf16 P staging; row stride 80B (16B-aligned, bank-spread)
    __shared__ __align__(16) unsigned char smp[8][TM][80];

    const int lane = threadIdx.x & 31;
    const int w    = threadIdx.x >> 5;
    const int tile = blockIdx.x * 8 + w;          // 0 .. 1023
    const int b    = tile >> 7;                   // 128 tiles per batch
    const int i0   = (tile & 127) * TM;

    const int* __restrict__ adjb = adj + (size_t)b * Nn * Nn;
    const float* __restrict__ t2b = t2 + b * Nn;

    float t1r[TM];
#pragma unroll
    for (int r = 0; r < TM; ++r) t1r[r] = t1[b * Nn + i0 + r];

    // ---------------- pass 1: row maxima (all e >= 0, so init 0 is exact) ---
    float mrow[TM];
#pragma unroll
    for (int r = 0; r < TM; ++r) mrow[r] = 0.f;

    for (int j0 = 0; j0 < Nn; j0 += BJ) {
        const int jj  = j0 + lane;
        const float t2v = t2b[jj];
        // sparse long-distance prefetch of the adj stream (1 KB ahead,
        // once every 8 j-blocks) -> global_prefetch_b8 without doubling
        // the vmem issue rate of the demand stream.
        if ((j0 & 255) == 0 && (j0 + 256 + BJ) <= Nn) {
#pragma unroll
            for (int r = 0; r < TM; ++r)
                __builtin_prefetch(&adjb[(size_t)(i0 + r) * Nn + jj + 256], 0, 0);
        }
#pragma unroll
        for (int r = 0; r < TM; ++r) {
            const int a = adjb[(size_t)(i0 + r) * Nn + jj];
            const float e = (a > 0) ? fmaxf(t1r[r] + t2v, 0.f) : 0.f;
            mrow[r] = fmaxf(mrow[r], e);
        }
    }
#pragma unroll
    for (int r = 0; r < TM; ++r)
#pragma unroll
        for (int off = 16; off >= 1; off >>= 1)
            mrow[r] = fmaxf(mrow[r], __shfl_xor(mrow[r], off, 32));

    // ---------------- pass 2: p = exp(e - m); acc += P(16x32) x hw(32x64) ---
    float lsum[TM];
#pragma unroll
    for (int r = 0; r < TM; ++r) lsum[r] = 0.f;
    v8f acc[4];
#pragma unroll
    for (int n = 0; n < 4; ++n) acc[n] = (v8f){};

    const unsigned short* __restrict__ hb = hwT + (size_t)b * Dd * Nn;
    const int off0 = (lane & 16);                  // A-operand byte offset per lane half

    for (int j0 = 0; j0 < Nn; j0 += BJ) {
        const int jj  = j0 + lane;
        const float t2v = t2b[jj];
#pragma unroll
        for (int r = 0; r < TM; ++r) {
            const int a = adjb[(size_t)(i0 + r) * Nn + jj];  // L2-resident reread
            const float e = (a > 0) ? fmaxf(t1r[r] + t2v, 0.f) : 0.f;
            const float p = __expf(e - mrow[r]);   // non-edges contribute exp(-m)
            lsum[r] += p;
            *(unsigned short*)&smp[w][r][2 * lane] = f2bf(p);
        }
        // order LDS stores (column layout) before transposed A-layout reads
        asm volatile("s_wait_dscnt 0" ::: "memory");

        // A operand (16x32 bf16): lane<16 -> K 0..7 / 16..23; lane>=16 -> +8
        V16 A;
        const unsigned char* rowp = &smp[w][lane & 15][0];
        A.q[0] = *(const uint4*)(rowp + off0);
        A.q[1] = *(const uint4*)(rowp + 32 + off0);
        asm volatile("" ::: "memory");             // keep next-iter stores after reads

        // Preload all four B operands (32x16 bf16 each) so the scheduler can
        // stagger s_wait_loadcnt and overlap loads with the WMMA chain.
        V16 Bv[4];
#pragma unroll
        for (int n = 0; n < 4; ++n) {
            const unsigned short* colp =
                hb + (size_t)(n * 16 + (lane & 15)) * Nn + j0 + ((lane & 16) >> 1);
            Bv[n].q[0] = *(const uint4*)(colp);
            Bv[n].q[1] = *(const uint4*)(colp + 16);
        }
#pragma unroll
        for (int n = 0; n < 4; ++n) {
            acc[n] = __builtin_amdgcn_wmma_f32_16x16x32_bf16(
                false, A.v, false, Bv[n].v, (short)0, acc[n], false, false);
        }
    }

    // ---------------- normalize + h_root select + store --------------------
#pragma unroll
    for (int r = 0; r < TM; ++r)
#pragma unroll
        for (int off = 16; off >= 1; off >>= 1)
            lsum[r] += __shfl_xor(lsum[r], off, 32);

    float inv[TM];
#pragma unroll
    for (int r = 0; r < TM; ++r) inv[r] = 1.0f / lsum[r];

    const bool hi = (lane & 16) != 0;
    const float* __restrict__ h0b = h0 + (size_t)(b * Nn + i0) * Dd;
    const float* __restrict__ hrb = h_root + b * Nn + i0;
    float* __restrict__ outb = out + (size_t)(b * Nn + i0) * Dd;

#pragma unroll
    for (int n = 0; n < 4; ++n) {
#pragma unroll
        for (int rr = 0; rr < 8; ++rr) {
            const int M   = rr + (hi ? 8 : 0);     // C/D layout: VGPR rr <-> row M
            const float s = hi ? inv[rr + 8] : inv[rr];
            const int col = n * 16 + (lane & 15);
            const float val = acc[n][rr] * s;
            const float hr  = hrb[M];
            const size_t o  = (size_t)M * Dd + col;
            outb[o] = (hr > 0.f) ? val : h0b[o];
        }
    }
}

extern "C" void kernel_launch(void* const* d_in, const int* in_sizes, int n_in,
                              void* d_out, int out_size, void* d_ws, size_t ws_size,
                              hipStream_t stream) {
    const float* h      = (const float*)d_in[0];
    const int*   adj    = (const int*)  d_in[1];
    const float* h_root = (const float*)d_in[2];
    const float* W      = (const float*)d_in[3];
    const float* a1     = (const float*)d_in[4];
    const float* a2     = (const float*)d_in[5];
    float* out = (float*)d_out;

    // workspace layout: hwT bf16 [8][64][2048] (2 MB), then t1, t2 (64 KB each)
    unsigned short* hwT = (unsigned short*)d_ws;
    float* t1 = (float*)((char*)d_ws + (size_t)Bq * Dd * Nn * 2);
    float* t2 = t1 + Bq * Nn;

    gat_prep_kernel<<<dim3((Bq * Nn) / 8), dim3(256), 0, stream>>>(
        h, W, a1, a2, hwT, t1, t2);
    gat_attn_kernel<<<dim3((Bq * (Nn / TM)) / 8), dim3(256), 0, stream>>>(
        h, adj, h_root, hwT, t1, t2, out);
}
